// EncLayer_84954453115392
// MI455X (gfx1250) — matrix operand order repaired and checked by reading, not compiled, not run on hardware
//
#include <hip/hip_runtime.h>
#include <hip/hip_bf16.h>
#include <math.h>

#define Bdim 2
#define Ndim 2048
#define Kdim 32
#define Hdim 256
#define H3   768
#define SCALE_INV (1.0f / 16.0f)
#define EPS 1e-5f

typedef _Float16 f16;
typedef __attribute__((ext_vector_type(16))) _Float16 v16h;
typedef __attribute__((ext_vector_type(8)))  _Float16 v8h;
typedef __attribute__((ext_vector_type(8)))  float    v8f;

union frag16 { uint4 u[2]; v16h v; };
union half8  { uint4 u;    v8h  v; };

__device__ __forceinline__ float gelu_exact(float x) {
    return 0.5f * x * (1.0f + erff(x * 0.70710678118654752f));
}

// A fragment: 16x32 f16 tile from LDS. Per-lane base pA = row base + kb
// (kb = 0 for lanes 0-15, 16 for lanes 16-31). Elements 0..7 <- pA[k0..k0+7],
// elements 8..15 <- pA[k0+8..k0+15]: two aligned 16B runs -> ds_load_b128 x2.
__device__ __forceinline__ v16h load_A(const f16* pA, int k0) {
    frag16 x;
    x.u[0] = *(const uint4*)(pA + k0);
    x.u[1] = *(const uint4*)(pA + k0 + 8);
    return x.v;
}

// B fragment: 32x16 f16 tile of W^T from global (W row-major [out,in]).
// Per-lane base pB = W + n*in_dim + kh (kh = 0/16 by lane half). Elements are
// contiguous in k: two aligned 16B runs -> global_load_b128 x2.
__device__ __forceinline__ v16h load_B(const f16* pB, int k0) {
    frag16 x;
    x.u[0] = *(const uint4*)(pB + k0);
    x.u[1] = *(const uint4*)(pB + k0 + 8);
    return x.v;
}

__device__ __forceinline__ v8h cvt8(const float* src) {
    float4 a = ((const float4*)src)[0];
    float4 b = ((const float4*)src)[1];
    v8h h;
    h[0] = (f16)a.x; h[1] = (f16)a.y; h[2] = (f16)a.z; h[3] = (f16)a.w;
    h[4] = (f16)b.x; h[5] = (f16)b.y; h[6] = (f16)b.z; h[7] = (f16)b.w;
    return h;
}

__global__ __launch_bounds__(256) void cvt_f32_f16(const float* __restrict__ s,
                                                   f16* __restrict__ d, int n) {
    int i = blockIdx.x * 256 + threadIdx.x;
    if (i < n) d[i] = (f16)s[i];
}

// One workgroup (8 wave32) per node: 32 edges x 3H concat -> 3-layer MLP via WMMA.
// IS_EDGE=0: out = LN1(h_V + sum_k m / 16)   [B*N, H]
// IS_EDGE=1: out = LN3(h_E + m)              [B*N*K, H]
template <int IS_EDGE>
__global__ __launch_bounds__(256)
void message_kernel(const float* __restrict__ hV, const float* __restrict__ hE,
                    const int* __restrict__ Eidx,
                    const f16* __restrict__ Wa, const float* __restrict__ ba,
                    const f16* __restrict__ Wb, const float* __restrict__ bb,
                    const f16* __restrict__ Wc, const float* __restrict__ bc,
                    const float* __restrict__ g, const float* __restrict__ be,
                    float* __restrict__ out) {
    __shared__ __align__(16) f16   sA[32 * 776];   // 32 x 768 concat tile (pad 8)
    __shared__ __align__(16) f16   sH1[32 * 264];  // layer1 act
    __shared__ __align__(16) f16   sH2[32 * 264];  // layer2 act
    __shared__ __align__(16) float sO[32 * 256];   // layer3 f32 out
    __shared__ float sRed[16];
    __shared__ int   sIdx[32];

    const int t    = threadIdx.x;
    const int lane = t & 31;
    const int wid  = t >> 5;
    const int node = blockIdx.x;            // b*N + n
    const int bat  = node / Ndim;

    if (t < 32) sIdx[t] = Eidx[node * Kdim + t];
    __syncthreads();

    // Stage concat [hv | he | hv_gather] as f16, 8 cols per thread-iter.
    for (int i = t; i < 32 * (H3 / 8); i += 256) {
        int r  = i / (H3 / 8);
        int cc = (i - r * (H3 / 8)) * 8;
        const float* src;
        if (cc < Hdim)            src = hV + node * Hdim + cc;
        else if (cc < 2 * Hdim)   src = hE + (node * Kdim + r) * Hdim + (cc - Hdim);
        else                      src = hV + (bat * Ndim + sIdx[r]) * Hdim + (cc - 2 * Hdim);
        *(v8h*)(sA + r * 776 + cc) = cvt8(src);
    }
    __syncthreads();

    const int row0 = (wid & 1) * 16;        // 2 row tiles
    const int cb   = (wid >> 1) * 4;        // 4 col tiles per wave (16 total)
    const int kb   = (lane & 16) ? 16 : 0;  // A-frag k sub-offset
    const int kh   = (lane & 16) ? 16 : 0;  // B-frag k half
    const int lm   = lane & 15;

    // ---- layer 1: [32,768] x [768,256], GELU -> sH1
    {
        const f16* pA = sA + (row0 + lm) * 776 + kb;
        const f16* pB0 = Wa + ((cb + 0) * 16 + lm) * H3 + kh;
        const f16* pB1 = Wa + ((cb + 1) * 16 + lm) * H3 + kh;
        const f16* pB2 = Wa + ((cb + 2) * 16 + lm) * H3 + kh;
        const f16* pB3 = Wa + ((cb + 3) * 16 + lm) * H3 + kh;
        v8f acc[4] = {};
#pragma unroll
        for (int k0 = 0; k0 < H3; k0 += 32) {
            v16h a = load_A(pA, k0);
            acc[0] = __builtin_amdgcn_wmma_f32_16x16x32_f16(false, a, false, load_B(pB0, k0), (short)0, acc[0], false, false);
            acc[1] = __builtin_amdgcn_wmma_f32_16x16x32_f16(false, a, false, load_B(pB1, k0), (short)0, acc[1], false, false);
            acc[2] = __builtin_amdgcn_wmma_f32_16x16x32_f16(false, a, false, load_B(pB2, k0), (short)0, acc[2], false, false);
            acc[3] = __builtin_amdgcn_wmma_f32_16x16x32_f16(false, a, false, load_B(pB3, k0), (short)0, acc[3], false, false);
        }
#pragma unroll
        for (int j = 0; j < 4; ++j) {
            int n = (cb + j) * 16 + lm;
            float bias = ba[n];
#pragma unroll
            for (int r = 0; r < 8; ++r) {
                int row = row0 + r + ((lane & 16) ? 8 : 0);
                sH1[row * 264 + n] = (f16)gelu_exact(acc[j][r] + bias);
            }
        }
    }
    __syncthreads();

    // ---- layer 2: [32,256] x [256,256], GELU -> sH2
    {
        const f16* pA = sH1 + (row0 + lm) * 264 + kb;
        const f16* pB0 = Wb + ((cb + 0) * 16 + lm) * Hdim + kh;
        const f16* pB1 = Wb + ((cb + 1) * 16 + lm) * Hdim + kh;
        const f16* pB2 = Wb + ((cb + 2) * 16 + lm) * Hdim + kh;
        const f16* pB3 = Wb + ((cb + 3) * 16 + lm) * Hdim + kh;
        v8f acc[4] = {};
#pragma unroll
        for (int k0 = 0; k0 < Hdim; k0 += 32) {
            v16h a = load_A(pA, k0);
            acc[0] = __builtin_amdgcn_wmma_f32_16x16x32_f16(false, a, false, load_B(pB0, k0), (short)0, acc[0], false, false);
            acc[1] = __builtin_amdgcn_wmma_f32_16x16x32_f16(false, a, false, load_B(pB1, k0), (short)0, acc[1], false, false);
            acc[2] = __builtin_amdgcn_wmma_f32_16x16x32_f16(false, a, false, load_B(pB2, k0), (short)0, acc[2], false, false);
            acc[3] = __builtin_amdgcn_wmma_f32_16x16x32_f16(false, a, false, load_B(pB3, k0), (short)0, acc[3], false, false);
        }
#pragma unroll
        for (int j = 0; j < 4; ++j) {
            int n = (cb + j) * 16 + lm;
            float bias = bb[n];
#pragma unroll
            for (int r = 0; r < 8; ++r) {
                int row = row0 + r + ((lane & 16) ? 8 : 0);
                sH2[row * 264 + n] = (f16)gelu_exact(acc[j][r] + bias);
            }
        }
    }
    __syncthreads();

    // ---- layer 3: [32,256] x [256,256] -> sO (f32); edge path adds residual here
    {
        const f16* pA = sH2 + (row0 + lm) * 264 + kb;
        const f16* pB0 = Wc + ((cb + 0) * 16 + lm) * Hdim + kh;
        const f16* pB1 = Wc + ((cb + 1) * 16 + lm) * Hdim + kh;
        const f16* pB2 = Wc + ((cb + 2) * 16 + lm) * Hdim + kh;
        const f16* pB3 = Wc + ((cb + 3) * 16 + lm) * Hdim + kh;
        v8f acc[4] = {};
#pragma unroll
        for (int k0 = 0; k0 < Hdim; k0 += 32) {
            v16h a = load_A(pA, k0);
            acc[0] = __builtin_amdgcn_wmma_f32_16x16x32_f16(false, a, false, load_B(pB0, k0), (short)0, acc[0], false, false);
            acc[1] = __builtin_amdgcn_wmma_f32_16x16x32_f16(false, a, false, load_B(pB1, k0), (short)0, acc[1], false, false);
            acc[2] = __builtin_amdgcn_wmma_f32_16x16x32_f16(false, a, false, load_B(pB2, k0), (short)0, acc[2], false, false);
            acc[3] = __builtin_amdgcn_wmma_f32_16x16x32_f16(false, a, false, load_B(pB3, k0), (short)0, acc[3], false, false);
        }
#pragma unroll
        for (int j = 0; j < 4; ++j) {
            int n = (cb + j) * 16 + lm;
            float bias = bc[n];
#pragma unroll
            for (int r = 0; r < 8; ++r) {
                int row = row0 + r + ((lane & 16) ? 8 : 0);
                float v = acc[j][r] + bias;
                if (IS_EDGE) v += hE[(node * Kdim + row) * Hdim + n];
                sO[row * 256 + n] = v;
            }
        }
    }
    __syncthreads();

    if (!IS_EDGE) {
        // sum over 32 edges, residual, block-wide LayerNorm over H=256
        float s = 0.f;
#pragma unroll 4
        for (int r = 0; r < 32; ++r) s += sO[r * 256 + t];
        float val = hV[node * Hdim + t] + s * SCALE_INV;
        float p = val;
#pragma unroll
        for (int o = 16; o > 0; o >>= 1) p += __shfl_xor(p, o, 32);
        if (lane == 0) sRed[wid] = p;
        __syncthreads();
        float mean = 0.f;
#pragma unroll
        for (int i = 0; i < 8; ++i) mean += sRed[i];
        mean *= (1.0f / 256.0f);
        float d = val - mean;
        float q = d * d;
#pragma unroll
        for (int o = 16; o > 0; o >>= 1) q += __shfl_xor(q, o, 32);
        if (lane == 0) sRed[8 + wid] = q;
        __syncthreads();
        float var = 0.f;
#pragma unroll
        for (int i = 0; i < 8; ++i) var += sRed[8 + i];
        var *= (1.0f / 256.0f);
        out[node * Hdim + t] = d * rsqrtf(var + EPS) * g[t] + be[t];
    } else {
        // per-edge LayerNorm: each wave handles 4 of the 32 rows
#pragma unroll
        for (int rr = 0; rr < 4; ++rr) {
            int row = wid * 4 + rr;
            float v[8];
            float s = 0.f;
#pragma unroll
            for (int i = 0; i < 8; ++i) { v[i] = sO[row * 256 + lane + 32 * i]; s += v[i]; }
#pragma unroll
            for (int o = 16; o > 0; o >>= 1) s += __shfl_xor(s, o, 32);
            float mean = s * (1.0f / 256.0f);
            float q = 0.f;
#pragma unroll
            for (int i = 0; i < 8; ++i) { float d = v[i] - mean; q += d * d; }
#pragma unroll
            for (int o = 16; o > 0; o >>= 1) q += __shfl_xor(q, o, 32);
            float inv = rsqrtf(q * (1.0f / 256.0f) + EPS);
#pragma unroll
            for (int i = 0; i < 8; ++i) {
                int c = lane + 32 * i;
                out[(node * Kdim + row) * Hdim + c] = (v[i] - mean) * inv * g[c] + be[c];
            }
        }
    }
}

// FFN: 32 nodes per workgroup: LN2(x + Wout(gelu(Win x + bin)) + bout)
__global__ __launch_bounds__(256)
void ffn_kernel(const float* __restrict__ x,
                const f16* __restrict__ Win,  const float* __restrict__ bin,
                const f16* __restrict__ Wout, const float* __restrict__ bout,
                const float* __restrict__ g,  const float* __restrict__ be,
                float* __restrict__ out) {
    __shared__ __align__(16) f16   sX[32 * 264];
    __shared__ __align__(16) f16   sH[32 * 520];
    __shared__ __align__(16) float sO[32 * 256];

    const int t = threadIdx.x, lane = t & 31, wid = t >> 5;
    const int nbase = blockIdx.x * 32;

    for (int i = t; i < 32 * (Hdim / 8); i += 256) {
        int r  = i >> 5;            // 32 chunks per row
        int cc = (i & 31) * 8;
        *(v8h*)(sX + r * 264 + cc) = cvt8(x + (nbase + r) * Hdim + cc);
    }
    __syncthreads();

    const int row0 = (wid & 1) * 16;
    const int kb   = (lane & 16) ? 16 : 0;
    const int kh   = (lane & 16) ? 16 : 0;
    const int lm   = lane & 15;

    // Win: [32,256] x [256,512], GELU -> sH  (8 col tiles per wave)
    {
        const int cb = (wid >> 1) * 8;
        const f16* pA = sX + (row0 + lm) * 264 + kb;
        const f16* pB[8];
#pragma unroll
        for (int j = 0; j < 8; ++j) pB[j] = Win + ((cb + j) * 16 + lm) * Hdim + kh;
        v8f acc[8] = {};
#pragma unroll
        for (int k0 = 0; k0 < Hdim; k0 += 32) {
            v16h a = load_A(pA, k0);
#pragma unroll
            for (int j = 0; j < 8; ++j)
                acc[j] = __builtin_amdgcn_wmma_f32_16x16x32_f16(false, a, false, load_B(pB[j], k0), (short)0, acc[j], false, false);
        }
#pragma unroll
        for (int j = 0; j < 8; ++j) {
            int n = (cb + j) * 16 + lm;
            float bias = bin[n];
#pragma unroll
            for (int r = 0; r < 8; ++r) {
                int row = row0 + r + ((lane & 16) ? 8 : 0);
                sH[row * 520 + n] = (f16)gelu_exact(acc[j][r] + bias);
            }
        }
    }
    __syncthreads();

    // Wout: [32,512] x [512,256] + bias + residual -> sO (4 col tiles per wave)
    {
        const int cb = (wid >> 1) * 4;
        const f16* pA = sH + (row0 + lm) * 520 + kb;
        const f16* pB0 = Wout + ((cb + 0) * 16 + lm) * (2 * Hdim) + kh;
        const f16* pB1 = Wout + ((cb + 1) * 16 + lm) * (2 * Hdim) + kh;
        const f16* pB2 = Wout + ((cb + 2) * 16 + lm) * (2 * Hdim) + kh;
        const f16* pB3 = Wout + ((cb + 3) * 16 + lm) * (2 * Hdim) + kh;
        v8f acc[4] = {};
#pragma unroll
        for (int k0 = 0; k0 < 2 * Hdim; k0 += 32) {
            v16h a = load_A(pA, k0);
            acc[0] = __builtin_amdgcn_wmma_f32_16x16x32_f16(false, a, false, load_B(pB0, k0), (short)0, acc[0], false, false);
            acc[1] = __builtin_amdgcn_wmma_f32_16x16x32_f16(false, a, false, load_B(pB1, k0), (short)0, acc[1], false, false);
            acc[2] = __builtin_amdgcn_wmma_f32_16x16x32_f16(false, a, false, load_B(pB2, k0), (short)0, acc[2], false, false);
            acc[3] = __builtin_amdgcn_wmma_f32_16x16x32_f16(false, a, false, load_B(pB3, k0), (short)0, acc[3], false, false);
        }
#pragma unroll
        for (int j = 0; j < 4; ++j) {
            int n = (cb + j) * 16 + lm;
            float bias = bout[n];
#pragma unroll
            for (int r = 0; r < 8; ++r) {
                int row = row0 + r + ((lane & 16) ? 8 : 0);
                sO[row * 256 + n] = acc[j][r] + bias + x[(nbase + row) * Hdim + n];
            }
        }
    }
    __syncthreads();

    // per-node LayerNorm (4 rows per wave)
#pragma unroll
    for (int rr = 0; rr < 4; ++rr) {
        int row = wid * 4 + rr;
        float v[8];
        float s = 0.f;
#pragma unroll
        for (int i = 0; i < 8; ++i) { v[i] = sO[row * 256 + lane + 32 * i]; s += v[i]; }
#pragma unroll
        for (int o = 16; o > 0; o >>= 1) s += __shfl_xor(s, o, 32);
        float mean = s * (1.0f / 256.0f);
        float q = 0.f;
#pragma unroll
        for (int i = 0; i < 8; ++i) { float d = v[i] - mean; q += d * d; }
#pragma unroll
        for (int o = 16; o > 0; o >>= 1) q += __shfl_xor(q, o, 32);
        float inv = rsqrtf(q * (1.0f / 256.0f) + EPS);
#pragma unroll
        for (int i = 0; i < 8; ++i) {
            int c = lane + 32 * i;
            out[(nbase + row) * Hdim + c] = (v[i] - mean) * inv * g[c] + be[c];
        }
    }
}

extern "C" void kernel_launch(void* const* d_in, const int* in_sizes, int n_in,
                              void* d_out, int out_size, void* d_ws, size_t ws_size,
                              hipStream_t stream) {
    const float* hV    = (const float*)d_in[0];
    const float* hE    = (const float*)d_in[1];
    const int*   Eidx  = (const int*)d_in[2];
    const float* W1_w  = (const float*)d_in[3];  const float* W1_b  = (const float*)d_in[4];
    const float* W2_w  = (const float*)d_in[5];  const float* W2_b  = (const float*)d_in[6];
    const float* W3_w  = (const float*)d_in[7];  const float* W3_b  = (const float*)d_in[8];
    const float* W11_w = (const float*)d_in[9];  const float* W11_b = (const float*)d_in[10];
    const float* W12_w = (const float*)d_in[11]; const float* W12_b = (const float*)d_in[12];
    const float* W13_w = (const float*)d_in[13]; const float* W13_b = (const float*)d_in[14];
    const float* Win_w = (const float*)d_in[15]; const float* Win_b = (const float*)d_in[16];
    const float* Wout_w= (const float*)d_in[17]; const float* Wout_b= (const float*)d_in[18];
    const float* n1_g  = (const float*)d_in[19]; const float* n1_b  = (const float*)d_in[20];
    const float* n2_g  = (const float*)d_in[21]; const float* n2_b  = (const float*)d_in[22];
    const float* n3_g  = (const float*)d_in[23]; const float* n3_b  = (const float*)d_in[24];

    // Workspace: f16 weights (L2-resident) + intermediate h_V1.
    f16* fW1   = (f16*)d_ws;
    f16* fW2   = fW1   + 256 * 768;
    f16* fW3   = fW2   + 256 * 256;
    f16* fW11  = fW3   + 256 * 256;
    f16* fW12  = fW11  + 256 * 768;
    f16* fW13  = fW12  + 256 * 256;
    f16* fWin  = fW13  + 256 * 256;
    f16* fWout = fWin  + 512 * 256;
    float* hV1 = (float*)(fWout + 256 * 512);

    auto cvt = [&](const float* s, f16* d, int n) {
        cvt_f32_f16<<<(n + 255) / 256, 256, 0, stream>>>(s, d, n);
    };
    cvt(W1_w,  fW1,  256 * 768);
    cvt(W2_w,  fW2,  256 * 256);
    cvt(W3_w,  fW3,  256 * 256);
    cvt(W11_w, fW11, 256 * 768);
    cvt(W12_w, fW12, 256 * 256);
    cvt(W13_w, fW13, 256 * 256);
    cvt(Win_w, fWin, 512 * 256);
    cvt(Wout_w,fWout,256 * 512);

    float* hV_out = (float*)d_out;
    float* hE_out = hV_out + Bdim * Ndim * Hdim;

    // Stage 1: node message + node update -> hV1
    message_kernel<0><<<Bdim * Ndim, 256, 0, stream>>>(
        hV, hE, Eidx, fW1, W1_b, fW2, W2_b, fW3, W3_b, n1_g, n1_b, hV1);
    // Stage 2: FFN + LN -> h_V output
    ffn_kernel<<<(Bdim * Ndim) / 32, 256, 0, stream>>>(
        hV1, fWin, Win_b, fWout, Wout_b, n2_g, n2_b, hV_out);
    // Stage 3: edge message (on updated h_V) + edge update -> h_E output
    message_kernel<1><<<Bdim * Ndim, 256, 0, stream>>>(
        hV_out, hE, Eidx, fW11, W11_b, fW12, W12_b, fW13, W13_b, n3_g, n3_b, hE_out);
}
